// SubgraphEmbeddingRegressorModel_34239479284341
// MI455X (gfx1250) — matrix operand order repaired and checked by reading, not compiled, not run on hardware
//
#include <hip/hip_runtime.h>
#include <hip/hip_bf16.h>

// ---------------- compile-time problem dims (match reference) ----------------
#define N_NODES   20000
#define NUM_DRUGS 2048
#define NUM_DPI   200000
#define E2        (2 * NUM_DPI)
#define IN_C      256
#define EMB       256
#define HID       512
#define N_CL      8
#define BATCH     16384
#define ROWS_PAD  (BATCH + 16 * N_CL)   // 16512
#define MAX_TILES (ROWS_PAD / 16)       // 1032

typedef __bf16 bf16_t;
typedef bf16_t v16bf __attribute__((ext_vector_type(16)));
typedef bf16_t v8bf  __attribute__((ext_vector_type(8)));
typedef float  v8f   __attribute__((ext_vector_type(8)));
typedef float  v4f   __attribute__((ext_vector_type(4)));

#define WMMA_BF16(A, B, C) \
    __builtin_amdgcn_wmma_f32_16x16x32_bf16(false, (A), false, (B), (short)0, (C), false, false)

// ---------------- init ----------------
__global__ void k_init(int* indeg, int* cursor, int* ccount, int* ccur, int* rows_sorted) {
    int i = blockIdx.x * blockDim.x + threadIdx.x;
    if (i < N_NODES) { indeg[i] = 0; cursor[i] = 0; }
    if (i < N_CL)    { ccount[i] = 0; ccur[i] = 0; }
    if (i < ROWS_PAD) rows_sorted[i] = -1;
}

// ---------------- B-operand pack: f32 [K,N] -> bf16 WMMA fragments ----------------
// dst[((kt*NT + nt)*32 + lane)*16 + i] = src[(kt*32 + koff(lane>>4, i))*N + nt*16 + (lane&15)]
__global__ void k_cvt_packB(const float* __restrict__ src, bf16_t* __restrict__ dst,
                            int K, int Nn) {
    int o = blockIdx.x * blockDim.x + threadIdx.x;
    int total = K * Nn;
    if (o >= total) return;
    int i    = o & 15;
    int lane = (o >> 4) & 31;
    int tile = o >> 9;
    int NT   = Nn >> 4;
    int nt = tile % NT;
    int kt = tile / NT;
    int k = kt * 32 + ((i < 8) ? ((lane >> 4) * 8 + i) : (16 + (lane >> 4) * 8 + (i - 8)));
    int n = nt * 16 + (lane & 15);
    dst[o] = (bf16_t)src[(size_t)k * Nn + n];
}

// ---------------- A-operand pack: f32 [M,K] -> bf16 WMMA A-fragments ----------------
// dst[((mt*KT + kt)*32 + lane)*16 + i] = src[(mt*16 + (lane&15))*K + kt*32 + koff(lane>>4, i)]
__global__ void k_cvt_packA(const float* __restrict__ src, bf16_t* __restrict__ dst,
                            int M, int K) {
    int o = blockIdx.x * blockDim.x + threadIdx.x;
    int total = M * K;
    if (o >= total) return;
    int i    = o & 15;
    int lane = (o >> 4) & 31;
    int frag = o >> 9;
    int KT   = K >> 5;
    int kt = frag % KT;
    int mt = frag / KT;
    int m = lane & 15, g = lane >> 4;
    int k = kt * 32 + ((i < 8) ? (g * 8 + i) : (16 + g * 8 + (i - 8)));
    int r = mt * 16 + m;
    dst[o] = (bf16_t)src[(size_t)r * K + k];
}

// ---------------- CSR build (by destination node) ----------------
__global__ void k_count(const int* __restrict__ col, int* __restrict__ indeg) {
    int e = blockIdx.x * blockDim.x + threadIdx.x;
    if (e < E2) atomicAdd(&indeg[col[e]], 1);
}

__global__ void k_scan(const int* __restrict__ indeg, int* __restrict__ csr_ptr) {
    __shared__ int part[1024];
    int tid = threadIdx.x;
    const int chunk = (N_NODES + 1023) / 1024;
    int base = tid * chunk;
    int end  = base + chunk; if (end > N_NODES) end = N_NODES;
    int s = 0;
    for (int i = base; i < end; ++i) s += indeg[i];
    part[tid] = s;
    __syncthreads();
    for (int off = 1; off < 1024; off <<= 1) {
        int v = (tid >= off) ? part[tid - off] : 0;
        __syncthreads();
        part[tid] += v;
        __syncthreads();
    }
    int run = part[tid] - s;  // exclusive prefix of this chunk
    for (int i = base; i < end; ++i) { csr_ptr[i] = run; run += indeg[i]; }
}

__global__ void k_dinv(const int* __restrict__ indeg, float* __restrict__ dinv) {
    int i = blockIdx.x * blockDim.x + threadIdx.x;
    if (i < N_NODES) dinv[i] = rsqrtf((float)(indeg[i] + 1));  // +1 self loop
}

__global__ void k_scatter(const int* __restrict__ row, const int* __restrict__ col,
                          const int* __restrict__ csr_ptr, int* __restrict__ cursor,
                          int* __restrict__ csr_src) {
    int e = blockIdx.x * blockDim.x + threadIdx.x;
    if (e < E2) {
        int c = col[e];
        int pos = atomicAdd(&cursor[c], 1);
        csr_src[csr_ptr[c] + pos] = row[e];
    }
}

// ---------------- dense GEMM: C[M,N] = Apacked[M,K] @ Wpacked[K,N] (bf16 -> f32) ----------
// one wave computes 32x64 (8 WMMA tiles); pointer-stepped, last-iteration-peeled pipeline
__global__ void __launch_bounds__(256) k_gemm(const bf16_t* __restrict__ Ap,
                                              const bf16_t* __restrict__ Wp,
                                              float* __restrict__ Cout,
                                              int M, int K, int Nn) {
    int lane = threadIdx.x & 31;
    int w    = threadIdx.x >> 5;
    int mblk = blockIdx.x * 8 + w;          // 32-row block
    if (mblk * 32 >= M) return;             // whole-wave exit (EXEC all-1 for WMMA)
    int col0 = blockIdx.y * 64;
    const int KT = K >> 5, NT = Nn >> 4;
    int mt0 = mblk * 2, mt1 = mt0 + 1;
    int m = lane & 15, g = lane >> 4;

    // fragment pointers at kt = 0 (each fragment = 512 elements = 1KB)
    const bf16_t* pa0 = Ap + (((size_t)mt0 * KT) * 32 + lane) * 16;
    const bf16_t* pa1 = Ap + (((size_t)mt1 * KT) * 32 + lane) * 16;
    const bf16_t* pb  = Wp + (((size_t)(col0 >> 4)) * 32 + lane) * 16;
    const size_t  stepA = 512;              // one kt step in A-pack
    const size_t  stepB = (size_t)NT * 512; // one kt step in B-pack

    v8f acc[8];
#pragma unroll
    for (int t = 0; t < 8; ++t)
#pragma unroll
        for (int j = 0; j < 8; ++j) acc[t][j] = 0.0f;

    v16bf a0 = *(const v16bf*)pa0;
    v16bf a1 = *(const v16bf*)pa1;
    v16bf b0 = *(const v16bf*)(pb);
    v16bf b1 = *(const v16bf*)(pb + 512);
    v16bf b2 = *(const v16bf*)(pb + 1024);
    v16bf b3 = *(const v16bf*)(pb + 1536);

    for (int kt = 0; kt < KT - 1; ++kt) {
        pa0 += stepA; pa1 += stepA; pb += stepB;
        v16bf a0n = *(const v16bf*)pa0;
        v16bf a1n = *(const v16bf*)pa1;
        v16bf b0n = *(const v16bf*)(pb);
        v16bf b1n = *(const v16bf*)(pb + 512);
        v16bf b2n = *(const v16bf*)(pb + 1024);
        v16bf b3n = *(const v16bf*)(pb + 1536);

        acc[0] = WMMA_BF16(a0, b0, acc[0]);
        acc[1] = WMMA_BF16(a0, b1, acc[1]);
        acc[2] = WMMA_BF16(a0, b2, acc[2]);
        acc[3] = WMMA_BF16(a0, b3, acc[3]);
        acc[4] = WMMA_BF16(a1, b0, acc[4]);
        acc[5] = WMMA_BF16(a1, b1, acc[5]);
        acc[6] = WMMA_BF16(a1, b2, acc[6]);
        acc[7] = WMMA_BF16(a1, b3, acc[7]);

        a0 = a0n; a1 = a1n; b0 = b0n; b1 = b1n; b2 = b2n; b3 = b3n;
    }
    // peeled last step: no prefetch
    acc[0] = WMMA_BF16(a0, b0, acc[0]);
    acc[1] = WMMA_BF16(a0, b1, acc[1]);
    acc[2] = WMMA_BF16(a0, b2, acc[2]);
    acc[3] = WMMA_BF16(a0, b3, acc[3]);
    acc[4] = WMMA_BF16(a1, b0, acc[4]);
    acc[5] = WMMA_BF16(a1, b1, acc[5]);
    acc[6] = WMMA_BF16(a1, b2, acc[6]);
    acc[7] = WMMA_BF16(a1, b3, acc[7]);

#pragma unroll
    for (int half = 0; half < 2; ++half)
#pragma unroll
        for (int t = 0; t < 4; ++t)
#pragma unroll
            for (int r = 0; r < 8; ++r) {
                int mi = (mt0 + half) * 16 + r + g * 8;
                Cout[(size_t)mi * Nn + col0 + t * 16 + m] = acc[half * 4 + t][r];
            }
}

// ---------------- GCN aggregation: relu(D^-1/2 A D^-1/2 T + b) ----------------
// Writes either f32 rows (Hout) or bf16 packed-A fragments (Apack) for the next GEMM.
__global__ void __launch_bounds__(256) k_agg(const float* __restrict__ T,
                                             const int* __restrict__ csr_ptr,
                                             const int* __restrict__ csr_src,
                                             const int* __restrict__ indeg,
                                             const float* __restrict__ dinv,
                                             const float* __restrict__ bias,
                                             float* __restrict__ Hout,
                                             bf16_t* __restrict__ Apack) {
    int w = threadIdx.x >> 5, lane = threadIdx.x & 31;
    int v = blockIdx.x * 8 + w;
    if (v >= N_NODES) return;
    int c0 = lane * 8;
    float s[8];
#pragma unroll
    for (int i = 0; i < 8; ++i) s[i] = 0.0f;
    int beg = csr_ptr[v], cnt = indeg[v];
    for (int e = 0; e < cnt; ++e) {
        int u = csr_src[beg + e];
        float du = dinv[u];
        const v4f* tp = (const v4f*)(T + (size_t)u * EMB + c0);
        v4f t0 = tp[0], t1 = tp[1];
#pragma unroll
        for (int i = 0; i < 4; ++i) { s[i] += t0[i] * du; s[4 + i] += t1[i] * du; }
    }
    float dv = dinv[v];
    const float* tv = T + (size_t)v * EMB + c0;
    float o[8];
#pragma unroll
    for (int i = 0; i < 8; ++i) {
        float val = s[i] * dv + tv[i] * (dv * dv) + bias[c0 + i];
        o[i] = fmaxf(val, 0.0f);
    }
    if (Hout) {
#pragma unroll
        for (int i = 0; i < 8; ++i) Hout[(size_t)v * EMB + c0 + i] = o[i];
    }
    if (Apack) {
        // lane's 8 channels = one contiguous 16B chunk of an A fragment
        const int KT = EMB >> 5;                     // 8
        int mt = v >> 4, mm = v & 15;
        int kt = lane >> 2;
        int fragLane = mm + ((lane & 1) << 4);
        int hi = (lane >> 1) & 1;
        v8bf pk;
#pragma unroll
        for (int i = 0; i < 8; ++i) pk[i] = (bf16_t)o[i];
        *(v8bf*)(Apack + ((((size_t)mt * KT + kt) * 32 + fragLane) << 4) + hi * 8) = pk;
    }
}

// ---------------- scatter-mean pool: graph[d] = mean(H[protein] over CSR[d]) ----------------
__global__ void __launch_bounds__(256) k_pool(const float* __restrict__ H,
                                              const int* __restrict__ csr_ptr,
                                              const int* __restrict__ csr_src,
                                              const int* __restrict__ indeg,
                                              float* __restrict__ graph) {
    int w = threadIdx.x >> 5, lane = threadIdx.x & 31;
    int d = blockIdx.x * 8 + w;
    if (d >= NUM_DRUGS) return;
    int c0 = lane * 8;
    float s[8];
#pragma unroll
    for (int i = 0; i < 8; ++i) s[i] = 0.0f;
    int beg = csr_ptr[d], cnt = indeg[d];
    for (int e = 0; e < cnt; ++e) {
        int p = csr_src[beg + e];
        const v4f* hp = (const v4f*)(H + (size_t)p * EMB + c0);
        v4f h0 = hp[0], h1 = hp[1];
#pragma unroll
        for (int i = 0; i < 4; ++i) { s[i] += h0[i]; s[4 + i] += h1[i]; }
    }
    float inv = 1.0f / (float)((cnt > 0) ? cnt : 1);
#pragma unroll
    for (int i = 0; i < 8; ++i) graph[(size_t)d * EMB + c0 + i] = s[i] * inv;
}

// ---------------- expert routing: counting sort of rows by cell line ----------------
__global__ void k_ccount(const int* __restrict__ ecl, int* __restrict__ ccount) {
    int r = blockIdx.x * blockDim.x + threadIdx.x;
    if (r < BATCH) atomicAdd(&ccount[ecl[r]], 1);
}

__global__ void k_coff(const int* __restrict__ ccount, int* __restrict__ coff,
                       int* __restrict__ tile_class) {
    if (threadIdx.x == 0 && blockIdx.x == 0) {
        int off = 0;
        for (int c = 0; c < N_CL; ++c) {
            coff[c] = off;
            int tl = (ccount[c] + 15) / 16;
            for (int t = 0; t < tl; ++t) tile_class[off / 16 + t] = c;
            off += tl * 16;                  // pad each class segment to 16 rows
        }
        for (int t = off / 16; t < MAX_TILES; ++t) tile_class[t] = 0;  // empty tail tiles
    }
}

__global__ void k_cscatter(const int* __restrict__ ecl, const int* __restrict__ coff,
                           int* __restrict__ ccur, int* __restrict__ rows_sorted) {
    int r = blockIdx.x * blockDim.x + threadIdx.x;
    if (r < BATCH) {
        int c = ecl[r];
        int pos = coff[c] + atomicAdd(&ccur[c], 1);
        rows_sorted[pos] = r;
    }
}

// ---------------- routed MoE MLP: out[r] = relu(xcat @ W1[c] + b1[c]) @ W2[c] + b2[c] ----------------
__global__ void __launch_bounds__(256) k_mlp(const float* __restrict__ graph,
                                             const int* __restrict__ dd,
                                             const int* __restrict__ rows_sorted,
                                             const int* __restrict__ tile_class,
                                             const bf16_t* __restrict__ W1p,
                                             const float* __restrict__ b1,
                                             const float* __restrict__ W2,
                                             const float* __restrict__ b2,
                                             float* __restrict__ outp) {
    __shared__ __attribute__((aligned(32))) bf16_t xs[16][2 * EMB];   // xcat tile, bf16
    __shared__ __attribute__((aligned(32))) bf16_t hs[16][HID];       // hidden tile, bf16
    __shared__ int    rows_s[16];
    __shared__ float  red[16][16];

    int tile = blockIdx.x;
    int tid  = threadIdx.x;
    int cl   = tile_class[tile];
    if (tid < 16) rows_s[tid] = rows_sorted[tile * 16 + tid];
    __syncthreads();

    // gather + convert xcat = [graph[ddA] , graph[ddB]] -> bf16 in LDS
    {
        int row = tid >> 4;
        int kb  = (tid & 15) * 32;
        int r   = rows_s[row];
        int dA  = (r >= 0) ? dd[r] : -1;
        int dB  = (r >= 0) ? dd[BATCH + r] : -1;
#pragma unroll
        for (int j = 0; j < 32; ++j) {
            int k = kb + j;
            float v = 0.0f;
            if (k < EMB) { if (dA >= 0) v = graph[(size_t)dA * EMB + k]; }
            else         { if (dB >= 0) v = graph[(size_t)dB * EMB + (k - EMB)]; }
            xs[row][k] = (bf16_t)v;
        }
    }
    __syncthreads();

    // GEMM1: hidden[16,512] = relu(xs @ W1[cl] + b1[cl]); pipelined, peeled
    {
        int lane = tid & 31, w = tid >> 5;
        int col0 = w * 64;
        int m = lane & 15, g = lane >> 4;
        const int NT = HID >> 4;              // 32
        const int KT = (2 * EMB) >> 5;        // 16
        const size_t stepB = (size_t)NT * 512;

        const bf16_t* pb = W1p + (size_t)cl * HID * HID + (((size_t)(col0 >> 4)) * 32 + lane) * 16;
        const bf16_t* pxlo = &xs[m][g * 8];
        const bf16_t* pxhi = &xs[m][16 + g * 8];

        auto ldA = [&](int koff) -> v16bf {
            v8bf alo = *(const v8bf*)(pxlo + koff);
            v8bf ahi = *(const v8bf*)(pxhi + koff);
            return __builtin_shufflevector(alo, ahi,
                                           0, 1, 2, 3, 4, 5, 6, 7,
                                           8, 9, 10, 11, 12, 13, 14, 15);
        };

        v8f acc[4];
#pragma unroll
        for (int t = 0; t < 4; ++t)
#pragma unroll
            for (int j = 0; j < 8; ++j) acc[t][j] = 0.0f;

        v16bf a  = ldA(0);
        v16bf b0 = *(const v16bf*)(pb);
        v16bf b1v = *(const v16bf*)(pb + 512);
        v16bf b2v = *(const v16bf*)(pb + 1024);
        v16bf b3v = *(const v16bf*)(pb + 1536);

        int koff = 0;
        for (int kt = 0; kt < KT - 1; ++kt) {
            koff += 32; pb += stepB;
            v16bf an  = ldA(koff);
            v16bf b0n = *(const v16bf*)(pb);
            v16bf b1n = *(const v16bf*)(pb + 512);
            v16bf b2n = *(const v16bf*)(pb + 1024);
            v16bf b3n = *(const v16bf*)(pb + 1536);
            acc[0] = WMMA_BF16(a, b0,  acc[0]);
            acc[1] = WMMA_BF16(a, b1v, acc[1]);
            acc[2] = WMMA_BF16(a, b2v, acc[2]);
            acc[3] = WMMA_BF16(a, b3v, acc[3]);
            a = an; b0 = b0n; b1v = b1n; b2v = b2n; b3v = b3n;
        }
        acc[0] = WMMA_BF16(a, b0,  acc[0]);
        acc[1] = WMMA_BF16(a, b1v, acc[1]);
        acc[2] = WMMA_BF16(a, b2v, acc[2]);
        acc[3] = WMMA_BF16(a, b3v, acc[3]);

#pragma unroll
        for (int t = 0; t < 4; ++t)
#pragma unroll
            for (int r = 0; r < 8; ++r) {
                int mi  = r + g * 8;
                int col = col0 + t * 16 + m;
                float val = acc[t][r] + b1[cl * HID + col];
                hs[mi][col] = (bf16_t)fmaxf(val, 0.0f);
            }
    }
    __syncthreads();

    // GEMM2 (N=1): out[r] = hs[row] . W2[cl] + b2[cl] ; 16 lanes reduce per row
    {
        int row = tid >> 4, seg = tid & 15;
        const float* w2 = W2 + (size_t)cl * HID;
        float s = 0.0f;
#pragma unroll
        for (int j = 0; j < 32; ++j) {
            int k = seg * 32 + j;
            s += (float)hs[row][k] * w2[k];
        }
        red[row][seg] = s;
    }
    __syncthreads();
    {
        int row = tid >> 4, seg = tid & 15;
        if (seg == 0) {
            int r = rows_s[row];
            if (r >= 0) {
                float tot = b2[cl];
#pragma unroll
                for (int t = 0; t < 16; ++t) tot += red[row][t];
                outp[r] = tot;
            }
        }
    }
}

// ---------------- host-side orchestration ----------------
extern "C" void kernel_launch(void* const* d_in, const int* in_sizes, int n_in,
                              void* d_out, int out_size, void* d_ws, size_t ws_size,
                              hipStream_t stream) {
    (void)in_sizes; (void)n_in; (void)out_size; (void)ws_size;
    const float* x    = (const float*)d_in[0];
    const int*   ei   = (const int*)d_in[1];   // [2, E2] row-major
    const int*   dd   = (const int*)d_in[2];   // [2, BATCH]
    const int*   ecl  = (const int*)d_in[3];   // [BATCH]
    const float* Wg1  = (const float*)d_in[5];
    const float* bg1  = (const float*)d_in[6];
    const float* Wg2  = (const float*)d_in[7];
    const float* bg2  = (const float*)d_in[8];
    const float* W1   = (const float*)d_in[9];
    const float* b1   = (const float*)d_in[10];
    const float* W2   = (const float*)d_in[11];
    const float* b2   = (const float*)d_in[12];
    float*       outp = (float*)d_out;

    const int* row = ei;
    const int* col = ei + E2;

    // carve workspace
    char* p = (char*)d_ws;
    auto carve = [&](size_t bytes) -> void* {
        void* r = (void*)p;
        p += (bytes + 255) & ~(size_t)255;
        return r;
    };
    int*    indeg       = (int*)carve((size_t)N_NODES * 4);
    int*    csr_ptr     = (int*)carve((size_t)N_NODES * 4);
    int*    cursor      = (int*)carve((size_t)N_NODES * 4);
    int*    csr_src     = (int*)carve((size_t)E2 * 4);
    float*  dinv        = (float*)carve((size_t)N_NODES * 4);
    int*    ccount      = (int*)carve(N_CL * 4);
    int*    coff        = (int*)carve(N_CL * 4);
    int*    ccur        = (int*)carve(N_CL * 4);
    int*    tile_class  = (int*)carve((size_t)MAX_TILES * 4);
    int*    rows_sorted = (int*)carve((size_t)ROWS_PAD * 4);
    float*  T           = (float*)carve((size_t)N_NODES * EMB * 4);
    float*  H           = (float*)carve((size_t)N_NODES * EMB * 4);
    float*  graph       = (float*)carve((size_t)NUM_DRUGS * EMB * 4);
    bf16_t* xAp         = (bf16_t*)carve((size_t)N_NODES * IN_C * 2);   // packed A (input x)
    bf16_t* hAp         = (bf16_t*)carve((size_t)N_NODES * EMB * 2);    // packed A (layer-1 act)
    bf16_t* Wg1bf       = (bf16_t*)carve((size_t)IN_C * EMB * 2);
    bf16_t* Wg2bf       = (bf16_t*)carve((size_t)EMB * EMB * 2);
    bf16_t* W1bf        = (bf16_t*)carve((size_t)N_CL * HID * HID * 2);

    // init + operand packing (WMMA fragment order, bf16)
    k_init<<<(N_NODES + 255) / 256, 256, 0, stream>>>(indeg, cursor, ccount, ccur, rows_sorted);
    k_cvt_packA<<<(N_NODES * IN_C + 255) / 256, 256, 0, stream>>>(x, xAp, N_NODES, IN_C);
    k_cvt_packB<<<(IN_C * EMB + 255) / 256, 256, 0, stream>>>(Wg1, Wg1bf, IN_C, EMB);
    k_cvt_packB<<<(EMB * EMB + 255) / 256, 256, 0, stream>>>(Wg2, Wg2bf, EMB, EMB);
    for (int c = 0; c < N_CL; ++c) {
        k_cvt_packB<<<(HID * HID + 255) / 256, 256, 0, stream>>>(
            W1 + (size_t)c * HID * HID, W1bf + (size_t)c * HID * HID, HID, HID);
    }

    // CSR build (by destination)
    k_count<<<(E2 + 255) / 256, 256, 0, stream>>>(col, indeg);
    k_scan<<<1, 1024, 0, stream>>>(indeg, csr_ptr);
    k_dinv<<<(N_NODES + 255) / 256, 256, 0, stream>>>(indeg, dinv);
    k_scatter<<<(E2 + 255) / 256, 256, 0, stream>>>(row, col, csr_ptr, cursor, csr_src);

    // GCN layer 1: T = x @ Wg1 ; hAp = packA(relu(agg(T) + bg1))
    dim3 gemm_grid((N_NODES / 32 + 7) / 8, EMB / 64);
    k_gemm<<<gemm_grid, 256, 0, stream>>>(xAp, Wg1bf, T, N_NODES, IN_C, EMB);
    k_agg<<<(N_NODES + 7) / 8, 256, 0, stream>>>(T, csr_ptr, csr_src, indeg, dinv, bg1,
                                                 nullptr, hAp);

    // GCN layer 2: T = h1 @ Wg2 ; H = relu(agg(T) + bg2)
    k_gemm<<<gemm_grid, 256, 0, stream>>>(hAp, Wg2bf, T, N_NODES, EMB, EMB);
    k_agg<<<(N_NODES + 7) / 8, 256, 0, stream>>>(T, csr_ptr, csr_src, indeg, dinv, bg2,
                                                 H, nullptr);

    // scatter-mean pool into per-drug embeddings (reuses CSR: cols < NUM_DRUGS)
    k_pool<<<(NUM_DRUGS + 7) / 8, 256, 0, stream>>>(H, csr_ptr, csr_src, indeg, graph);

    // route rows to experts, then run routed MLP
    k_ccount<<<(BATCH + 255) / 256, 256, 0, stream>>>(ecl, ccount);
    k_coff<<<1, 1, 0, stream>>>(ccount, coff, tile_class);
    k_cscatter<<<(BATCH + 255) / 256, 256, 0, stream>>>(ecl, coff, ccur, rows_sorted);
    k_mlp<<<MAX_TILES, 256, 0, stream>>>(graph, dd, rows_sorted, tile_class,
                                         W1bf, b1, W2, b2, outp);
}
